// Head_30786325577999
// MI455X (gfx1250) — matrix operand order repaired and checked by reading, not compiled, not run on hardware
//
#include <hip/hip_runtime.h>
#include <hip/hip_bf16.h>

typedef __attribute__((ext_vector_type(16))) _Float16 v16h;
typedef __attribute__((ext_vector_type(2)))  _Float16 h2v;
typedef __attribute__((ext_vector_type(8)))  float    v8f;

#define B_SZ 512
#define T_SZ 256
#define C_SZ 384
#define D_SZ 64

static __device__ __forceinline__ v8f vzero8() {
    v8f z;
#pragma unroll
    for (int i = 0; i < 8; ++i) z[i] = 0.0f;
    return z;
}

// A-fragment K index base (16-bit A 16x32, ISA 7.12.2): K = 16*(v/4) + 2*(v%4) + 8*g + slot
static __device__ __forceinline__ int kbaseA(int v, int g) {
    return 16 * (v >> 2) + 2 * (v & 3) + 8 * g;
}
// B-fragment K index base (16-bit B 32x16): K = 16*(v/4) + 2*(v%4) + 8*g + slot
static __device__ __forceinline__ int kbaseB(int v, int g) {
    return 16 * (v >> 2) + 2 * (v & 3) + 8 * g;
}

// ---------------------------------------------------------------------------
// Prep: convert Wq/Wk/Wv (f32 [384][64]) into f16 B-fragment register order.
// Layout: wf[mat][chunk(12)][ntile(4)][lane(32)][half(16)]
// ---------------------------------------------------------------------------
__global__ __launch_bounds__(32) void prep_w_kernel(const float* __restrict__ Wq,
                                                    const float* __restrict__ Wk,
                                                    const float* __restrict__ Wv,
                                                    _Float16* __restrict__ wf) {
    const int bid = blockIdx.x;          // 0..143 = mat*48 + chunk*4 + ntile
    const int mat = bid / 48;
    const int rem = bid % 48;
    const int chunk = rem / 4;
    const int j = rem % 4;
    const float* W = (mat == 0) ? Wq : (mat == 1) ? Wk : Wv;
    const int lane = threadIdx.x & 31;
    const int g = lane >> 4;
    const int n = lane & 15;
    _Float16* dst = wf + (((size_t)bid * 32) + lane) * 16;
#pragma unroll
    for (int h = 0; h < 16; ++h) {
        const int v = h >> 1, slot = h & 1;
        const int k = kbaseB(v, g) + slot;
        const int c = chunk * 32 + k;
        const int d = j * 16 + n;
        dst[h] = (_Float16)W[c * D_SZ + d];
    }
}

// ---------------------------------------------------------------------------
// Fused attention head: one workgroup (8 waves) per batch element.
// ---------------------------------------------------------------------------
__global__ __launch_bounds__(256) void attn_head_kernel(const float* __restrict__ x,
                                                        const _Float16* __restrict__ wf,
                                                        float* __restrict__ out) {
    __shared__ _Float16 lds_q[T_SZ * D_SZ];    // 32 KB, row-major [t][d]
    __shared__ _Float16 lds_k[T_SZ * D_SZ];    // 32 KB, row-major [s][d]
    __shared__ _Float16 lds_vt[D_SZ * T_SZ];   // 32 KB, transposed [d][s]
    __shared__ _Float16 lds_p[8][16 * 32];     // 8 KB, per-wave P staging [m][k]

    const int b    = blockIdx.x;
    const int tid  = threadIdx.x;
    const int wv   = tid >> 5;
    const int lane = tid & 31;
    const int g    = lane >> 4;
    const int ln   = lane & 15;

    const float* xb = x + (size_t)b * T_SZ * C_SZ;

    // ---------------- Phase 1: Q/K/V projection into LDS ----------------
#pragma unroll
    for (int tt = 0; tt < 2; ++tt) {
        const int tile = wv * 2 + tt;
        const int t0 = tile * 16;
        v8f accq[4], acck[4], accv[4];
#pragma unroll
        for (int j = 0; j < 4; ++j) { accq[j] = vzero8(); acck[j] = vzero8(); accv[j] = vzero8(); }

        const float* xrow = xb + (size_t)(t0 + ln) * C_SZ;  // A: lane holds row M=ln
#pragma unroll
        for (int ch = 0; ch < 12; ++ch) {
            const int c0 = ch * 32;
            // A fragment: x[t0+ln][c0 + K_A], f32 -> f16
            v16h a;
#pragma unroll
            for (int v = 0; v < 8; ++v) {
                const float2 f = *(const float2*)(xrow + c0 + kbaseA(v, g));
                a[2 * v]     = (_Float16)f.x;
                a[2 * v + 1] = (_Float16)f.y;
            }
#pragma unroll
            for (int mat = 0; mat < 3; ++mat) {
#pragma unroll
                for (int j = 0; j < 4; ++j) {
                    const v16h bf = *(const v16h*)(wf + ((((size_t)mat * 12 + ch) * 4 + j) * 32 + lane) * 16);
                    v8f& acc = (mat == 0 ? accq : (mat == 1 ? acck : accv))[j];
                    acc = __builtin_amdgcn_wmma_f32_16x16x32_f16(false, a, false, bf,
                                                                 (short)0, acc, false, false);
                }
            }
        }
        // Scatter D-tiles to LDS (Q,K row-major; V transposed), f32 -> f16.
#pragma unroll
        for (int j = 0; j < 4; ++j) {
#pragma unroll
            for (int r = 0; r < 8; ++r) {
                const int m = r + 8 * g;
                const int t = t0 + m;
                const int d = j * 16 + ln;
                lds_q[t * D_SZ + d]  = (_Float16)accq[j][r];
                lds_k[t * D_SZ + d]  = (_Float16)acck[j][r];
                lds_vt[d * T_SZ + t] = (_Float16)accv[j][r];
            }
        }
    }
    __syncthreads();

    // ---------------- Phase 2: causal softmax attention ----------------
    const float scale = 0.125f;  // 1/sqrt(64)
#pragma unroll
    for (int qi = 0; qi < 2; ++qi) {
        const int qt = qi ? (15 - wv) : wv;   // pairing {w, 15-w}: balanced causal work
        const int t0 = qt * 16;

        // Q A-fragments for d-chunks 0..31 and 32..63
        v16h aq[2];
#pragma unroll
        for (int dc = 0; dc < 2; ++dc) {
#pragma unroll
            for (int v = 0; v < 8; ++v) {
                const h2v q2 = *(const h2v*)&lds_q[(t0 + ln) * D_SZ + dc * 32 + kbaseA(v, g)];
                aq[dc][2 * v]     = q2[0];
                aq[dc][2 * v + 1] = q2[1];
            }
        }

        // S = Q K^T * scale, with causal mask; keep all 16 s-tiles in registers.
        v8f sc[16];
#pragma unroll
        for (int st = 0; st < 16; ++st) {
            if (st <= qt) {
                v8f acc = vzero8();
#pragma unroll
                for (int dc = 0; dc < 2; ++dc) {
                    v16h bk;
#pragma unroll
                    for (int v = 0; v < 8; ++v) {
                        const h2v k2 = *(const h2v*)&lds_k[(st * 16 + ln) * D_SZ + dc * 32 + kbaseB(v, g)];
                        bk[2 * v]     = k2[0];
                        bk[2 * v + 1] = k2[1];
                    }
                    acc = __builtin_amdgcn_wmma_f32_16x16x32_f16(false, aq[dc], false, bk,
                                                                 (short)0, acc, false, false);
                }
#pragma unroll
                for (int r = 0; r < 8; ++r) {
                    float s = acc[r] * scale;
                    if (st == qt && ln > r + 8 * g) s = -1e30f;  // diagonal tile mask: n > m
                    sc[st][r] = s;
                }
            } else {
#pragma unroll
                for (int r = 0; r < 8; ++r) sc[st][r] = -1e30f;
            }
        }

        // Row-wise softmax: rows live across the 16 lanes of each half-wave.
        float lrow[8];
#pragma unroll
        for (int r = 0; r < 8; ++r) {
            float m = -1e30f;
#pragma unroll
            for (int st = 0; st < 16; ++st) m = fmaxf(m, sc[st][r]);
#pragma unroll
            for (int s = 1; s < 16; s <<= 1) m = fmaxf(m, __shfl_xor(m, s, 16));
            float l = 0.0f;
#pragma unroll
            for (int st = 0; st < 16; ++st) {
                const float p = __expf(sc[st][r] - m);
                sc[st][r] = p;   // unnormalized probs; 1/l folded into epilogue
                l += p;
            }
#pragma unroll
            for (int s = 1; s < 16; s <<= 1) l += __shfl_xor(l, s, 16);
            lrow[r] = l;
        }

        // O = P V : stage P (D-layout) through per-wave LDS to build A-fragments.
        v8f o[4];
#pragma unroll
        for (int j = 0; j < 4; ++j) o[j] = vzero8();
        _Float16* pb = lds_p[wv];
#pragma unroll
        for (int c = 0; c < 8; ++c) {
            if (2 * c <= qt) {
#pragma unroll
                for (int u = 0; u < 2; ++u) {
                    const int st = 2 * c + u;
#pragma unroll
                    for (int r = 0; r < 8; ++r) {
                        const int m = r + 8 * g;
                        pb[m * 32 + u * 16 + ln] = (_Float16)sc[st][r];
                    }
                }
                // Cross-lane RAW through LDS within the wave: DS is in-order, but
                // keep the compiler from reordering and drain DScnt explicitly.
                asm volatile("s_wait_dscnt 0" ::: "memory");
                v16h ap;
#pragma unroll
                for (int v = 0; v < 8; ++v) {
                    const h2v p2 = *(const h2v*)&pb[ln * 32 + kbaseA(v, g)];
                    ap[2 * v]     = p2[0];
                    ap[2 * v + 1] = p2[1];
                }
#pragma unroll
                for (int j = 0; j < 4; ++j) {
                    v16h bvf;
#pragma unroll
                    for (int v = 0; v < 8; ++v) {
                        const h2v v2 = *(const h2v*)&lds_vt[(j * 16 + ln) * T_SZ + 32 * c + kbaseB(v, g)];
                        bvf[2 * v]     = v2[0];
                        bvf[2 * v + 1] = v2[1];
                    }
                    o[j] = __builtin_amdgcn_wmma_f32_16x16x32_f16(false, ap, false, bvf,
                                                                  (short)0, o[j], false, false);
                }
                // WAR guard before next iteration restages pb.
                asm volatile("s_wait_dscnt 0" ::: "memory");
            }
        }

        // Epilogue: out[b, t0+m, d] = o * (1/l)
        float* ob = out + ((size_t)b * T_SZ + t0) * D_SZ;
#pragma unroll
        for (int r = 0; r < 8; ++r) {
            const float inv = 1.0f / lrow[r];
#pragma unroll
            for (int j = 0; j < 4; ++j) {
                ob[(r + 8 * g) * D_SZ + j * 16 + ln] = o[j][r] * inv;
            }
        }
    }
}

extern "C" void kernel_launch(void* const* d_in, const int* in_sizes, int n_in,
                              void* d_out, int out_size, void* d_ws, size_t ws_size,
                              hipStream_t stream) {
    const float* x  = (const float*)d_in[0];
    const float* Wq = (const float*)d_in[1];
    const float* Wk = (const float*)d_in[2];
    const float* Wv = (const float*)d_in[3];
    _Float16* wf = (_Float16*)d_ws;   // 3*12*4*32*16 halves = 147456 bytes

    prep_w_kernel<<<144, 32, 0, stream>>>(Wq, Wk, Wv, wf);
    attn_head_kernel<<<B_SZ, 256, 0, stream>>>(x, wf, (float*)d_out);
}
